// CausalGraphNetwork_86320252715296
// MI455X (gfx1250) — matrix operand order repaired and checked by reading, not compiled, not run on hardware
//
#include <hip/hip_runtime.h>
#include <math.h>

typedef _Float16 v8h  __attribute__((ext_vector_type(8)));
typedef _Float16 v16h __attribute__((ext_vector_type(16)));
typedef float    v8f  __attribute__((ext_vector_type(8)));

#define BATCH 2
#define SEQ   512
#define HDIM  256
#define MROWS (BATCH * SEQ)   // 1024

// ---------------------------------------------------------------------------
// f32 -> f16 conversion
__global__ void cvt_f32_f16(const float* __restrict__ src,
                            _Float16* __restrict__ dst, int n) {
    int i = blockIdx.x * 256 + threadIdx.x;
    if (i < n) dst[i] = (_Float16)src[i];
}

// Extract Wa (off=0) or Wb (off=HDIM) from W_ep1 (HDIM x 2*HDIM, row-major),
// converting to f16. dst is HDIM x HDIM row-major.
__global__ void extract_ep1(const float* __restrict__ W_ep1,
                            _Float16* __restrict__ dst, int off) {
    int idx = blockIdx.x * 256 + threadIdx.x;   // HDIM*HDIM threads
    int k = idx >> 8;
    int h = idx & 255;
    dst[idx] = (_Float16)W_ep1[k * (2 * HDIM) + off + h];
}

// ---------------------------------------------------------------------------
// out = act(A @ W^T + bias), A: M x K (f16), W: N x K row-major (f16),
// out: M x N (f16).  M=1024, N=256, K=256 hard-coded.
// Block = 128 threads (4 waves). Block tile = 64(M) x 64(N).
// Each wave: 16(M) x 64(N) via 4 accumulators, K-loop of 8 x wmma(16x16x32).
__global__ void gemm_wmma(const _Float16* __restrict__ A,
                          const _Float16* __restrict__ W,
                          const float* __restrict__ bias,
                          _Float16* __restrict__ out,
                          int do_relu) {
    const int lane = threadIdx.x & 31;
    const int wave = threadIdx.x >> 5;            // 0..3
    const int sel  = lane >> 4;                   // 0 or 1
    const int lr   = lane & 15;

    const int m0 = blockIdx.x * 64 + wave * 16;   // wave's M tile
    const int n0 = blockIdx.y * 64;               // block's N tile (4 sub-tiles)

    v8f acc[4] = {};

    const _Float16* arow = A + (size_t)(m0 + lr) * HDIM;

    #pragma unroll
    for (int k0 = 0; k0 < HDIM; k0 += 32) {
        // --- A fragment (16-bit A 16x32 layout):
        // lane<16: k in {0..15}; lane>=16: {16..31} (contiguous, matches B)
        const _Float16* pa = arow + k0 + sel * 16;
        v8h lo = *(const v8h*)pa;
        v8h hi = *(const v8h*)(pa + 8);
        v16h av;
        #pragma unroll
        for (int e = 0; e < 8; ++e) { av[e] = lo[e]; av[e + 8] = hi[e]; }

        #pragma unroll
        for (int nt = 0; nt < 4; ++nt) {
            // --- B fragment (B is 32x16; row n striped across lanes):
            // lane holds column n = n0+nt*16+lr, k = k0 + sel*16 + (0..15) contiguous
            const _Float16* pb =
                W + (size_t)(n0 + nt * 16 + lr) * HDIM + k0 + sel * 16;
            v16h bv = *(const v16h*)pb;
            acc[nt] = __builtin_amdgcn_wmma_f32_16x16x32_f16(
                false, av, false, bv, (short)0, acc[nt], false, false);
        }
    }

    // --- store: lane l vgpr r -> D[m0 + r + 8*sel][n0 + nt*16 + lr]
    #pragma unroll
    for (int nt = 0; nt < 4; ++nt) {
        const int col = n0 + nt * 16 + lr;
        const float bi = bias ? bias[col] : 0.0f;
        #pragma unroll
        for (int r = 0; r < 8; ++r) {
            int row = m0 + r + 8 * sel;
            float v = acc[nt][r] + bi;
            if (do_relu) v = fmaxf(v, 0.0f);
            out[(size_t)row * HDIM + col] = (_Float16)v;
        }
    }
}

// ---------------------------------------------------------------------------
// scores[b,i,j] = sigmoid( sum_h relu(a[b,i,h]+c[b,j,h]+b1[h]) * w2[h] + b2 )
//                * (j < i)
// One block per 16x16 (i,j) tile per batch. a/c rows staged in LDS as f32.
__global__ void edge_score(const _Float16* __restrict__ Ah,
                           const _Float16* __restrict__ Ch,
                           const float* __restrict__ b_ep1,
                           const float* __restrict__ w_ep2,
                           const float* __restrict__ b_ep2,
                           float* __restrict__ out) {
    const int j0 = blockIdx.x * 16;
    const int i0 = blockIdx.y * 16;
    const int b  = blockIdx.z;
    const int tx = threadIdx.x;      // j within tile
    const int ty = threadIdx.y;      // i within tile
    const int i  = i0 + ty;
    const int j  = j0 + tx;
    const size_t oidx = ((size_t)b * SEQ + i) * SEQ + j;

    // Tile entirely masked (j >= i for every pair) -> just write zeros.
    if (j0 >= i0 + 15) { out[oidx] = 0.0f; return; }

    __shared__ float aS[16][260];    // stride 260: conflict-free column reads
    __shared__ float cS[16][260];
    __shared__ float bS[HDIM];
    __shared__ float wS[HDIM];

    const int tid = ty * 16 + tx;    // 0..255
    {
        const int r = tid >> 4;            // row within tile
        const int h = (tid & 15) * 16;     // 16 elements per thread
        const _Float16* pa = Ah + ((size_t)(b * SEQ + i0 + r)) * HDIM + h;
        const _Float16* pc = Ch + ((size_t)(b * SEQ + j0 + r)) * HDIM + h;
        #pragma unroll
        for (int e = 0; e < 16; ++e) {
            aS[r][h + e] = (float)pa[e];
            cS[r][h + e] = (float)pc[e];
        }
        bS[tid] = b_ep1[tid];
        wS[tid] = w_ep2[tid];
    }
    __syncthreads();

    float accv = 0.0f;
    #pragma unroll 8
    for (int h = 0; h < HDIM; ++h) {
        float t = aS[ty][h] + cS[tx][h] + bS[h];
        accv = fmaf(fmaxf(t, 0.0f), wS[h], accv);
    }

    float s = accv + b_ep2[0];
    s = 1.0f / (1.0f + expf(-s));
    out[oidx] = (j < i) ? s : 0.0f;
}

// ---------------------------------------------------------------------------
extern "C" void kernel_launch(void* const* d_in, const int* in_sizes, int n_in,
                              void* d_out, int out_size, void* d_ws, size_t ws_size,
                              hipStream_t stream) {
    const float* X      = (const float*)d_in[0];   // (B,N,H)
    // d_in[1] = step_mask: unused by the reference
    const float* W_gc1  = (const float*)d_in[2];   // (H,H)
    const float* b_gc1  = (const float*)d_in[3];   // (H,)
    const float* W_gc2  = (const float*)d_in[4];   // (H,H)
    const float* b_gc2  = (const float*)d_in[5];   // (H,)
    const float* W_ep1  = (const float*)d_in[6];   // (H,2H)
    const float* b_ep1  = (const float*)d_in[7];   // (H,)
    const float* w_ep2  = (const float*)d_in[8];   // (H,)
    const float* b_ep2  = (const float*)d_in[9];   // scalar
    float* out = (float*)d_out;                    // (B,N,N)

    const int act = MROWS * HDIM;                  // 262144
    const int wsz = HDIM * HDIM;                   // 65536

    _Float16* Xh  = (_Float16*)d_ws;
    _Float16* Hh  = Xh  + act;
    _Float16* Fh  = Hh  + act;
    _Float16* Ah  = Fh  + act;
    _Float16* Ch  = Ah  + act;
    _Float16* W1h = Ch  + act;
    _Float16* W2h = W1h + wsz;
    _Float16* Wah = W2h + wsz;
    _Float16* Wbh = Wah + wsz;
    (void)ws_size; (void)n_in; (void)in_sizes; (void)out_size;

    // --- precision prep: everything into f16 for the WMMA path
    cvt_f32_f16<<<(act + 255) / 256, 256, 0, stream>>>(X, Xh, act);
    cvt_f32_f16<<<(wsz + 255) / 256, 256, 0, stream>>>(W_gc1, W1h, wsz);
    cvt_f32_f16<<<(wsz + 255) / 256, 256, 0, stream>>>(W_gc2, W2h, wsz);
    extract_ep1<<<(wsz + 255) / 256, 256, 0, stream>>>(W_ep1, Wah, 0);
    extract_ep1<<<(wsz + 255) / 256, 256, 0, stream>>>(W_ep1, Wbh, HDIM);

    // --- 4 WMMA GEMMs (M=1024, N=256, K=256)
    dim3 ggrid(MROWS / 64, HDIM / 64);
    gemm_wmma<<<ggrid, 128, 0, stream>>>(Xh, W1h, b_gc1, Hh, 1);
    gemm_wmma<<<ggrid, 128, 0, stream>>>(Hh, W2h, b_gc2, Fh, 1);
    gemm_wmma<<<ggrid, 128, 0, stream>>>(Fh, Wah, nullptr, Ah, 0);
    gemm_wmma<<<ggrid, 128, 0, stream>>>(Fh, Wbh, nullptr, Ch, 0);

    // --- fused pairwise edge scoring (never materializes the BxNxNxH tensor)
    dim3 egrid(SEQ / 16, SEQ / 16, BATCH);
    edge_score<<<egrid, dim3(16, 16), 0, stream>>>(Ah, Ch, b_ep1, w_ep2, b_ep2, out);
}
